// GroupedAttention_17033840296421
// MI455X (gfx1250) — compile-verified, matrix operand order checked by reading, not run on hardware
//
#include <hip/hip_runtime.h>
#include <hip/hip_bf16.h>
#include <stdint.h>

#define SEQ  2048
#define BSZ  2
#define DIM  256
#define NG   2
#define NSUB 4
#define NTOK (SEQ*BSZ)          // 4096
#define KD   (DIM*NG*NSUB)      // 2048 (out-proj contraction width)

typedef __bf16 bf16_t;
typedef __bf16 bf16x16 __attribute__((ext_vector_type(16)));
typedef float  f32x8   __attribute__((ext_vector_type(8)));
typedef float  f32x4   __attribute__((ext_vector_type(4)));
typedef uint32_t u32x4 __attribute__((ext_vector_type(4)));
typedef uint32_t u32x2 __attribute__((ext_vector_type(2)));

union Frag {
    bf16x16 v;
    u32x4   q[2];
};

__device__ __forceinline__ uint32_t bfbits(float f) {
    union { float f; uint32_t u; } a; a.f = f;
    return (a.u + 0x7FFFu + ((a.u >> 16) & 1u)) >> 16;      // RNE
}
__device__ __forceinline__ uint32_t pk2(float lo, float hi) {
    return bfbits(lo) | (bfbits(hi) << 16);
}
__device__ __forceinline__ bf16_t f2bf(float f) {
    union { uint16_t s; bf16_t b; } o; o.s = (uint16_t)bfbits(f);
    return o.b;
}

__device__ __forceinline__ f32x8 wmma_bf16(const Frag& a, const Frag& b, f32x8 c) {
    return __builtin_amdgcn_wmma_f32_16x16x32_bf16(false, a.v, false, b.v,
                                                   (short)0, c, false, false);
}

// Stage a 32(k) x 256(n) fp32 tile into LDS as bf16, transposed [n][k].
// 4x4 units: 4 coalesced float4 loads -> packed b64 LDS stores.
__device__ __forceinline__ void stage_B(bf16_t* __restrict__ Blds,
                                        const float* __restrict__ W,
                                        int kb, int ldw, int cb, int t)
{
#pragma unroll
    for (int u = 0; u < 4; ++u) {
        const int uid = t + u*128;           // 0..511
        const int uk  = uid >> 6;            // k unit 0..7
        const int un  = uid & 63;            // n unit 0..63
        const float* wp = W + (size_t)(kb + uk*4)*ldw + cb + un*4;
        const f32x4 w0 = *(const f32x4*)(wp);
        const f32x4 w1 = *(const f32x4*)(wp + ldw);
        const f32x4 w2 = *(const f32x4*)(wp + 2*ldw);
        const f32x4 w3 = *(const f32x4*)(wp + 3*ldw);
#pragma unroll
        for (int nn = 0; nn < 4; ++nn) {
            u32x2 p;
            p[0] = pk2(w0[nn], w1[nn]);
            p[1] = pk2(w2[nn], w3[nn]);
            *(u32x2*)&Blds[(un*4 + nn)*32 + uk*4] = p;
        }
    }
}

// ---------------------------------------------------------------------------
// Kernel 1: fused QKV projection + bias + LayerNorm(K,Q) + bf16 cast.
// Block = 128 threads (4 waves), computes a 16-row x 256-col head segment.
// Segments: 0-1 = K(g), 2-3 = V(g) (stored transposed), 4-11 = Q(g,sub).
// ---------------------------------------------------------------------------
__global__ __launch_bounds__(128)
void proj_qkv_ln(const float* __restrict__ x,
                 const float* __restrict__ WKw, const float* __restrict__ WKb,
                 const float* __restrict__ WVw, const float* __restrict__ WVb,
                 const float* __restrict__ WQw, const float* __restrict__ WQb,
                 const float* __restrict__ lng, const float* __restrict__ lnb,
                 bf16_t* __restrict__ Kn, bf16_t* __restrict__ Qn,
                 bf16_t* __restrict__ Vt)
{
    __shared__ __align__(16) bf16_t Alds[16*32];
    __shared__ __align__(16) bf16_t Blds[256*32];      // transposed: [n][k]
    __shared__ __align__(16) float  Ct[16*256];
    __shared__ float psum[16*8], psq[16*8];
    __shared__ float rmean[16], rrstd[16];

    const int t    = threadIdx.x;
    const int lane = t & 31, wv = t >> 5;
    const int half = lane >> 4, ln16 = lane & 15;
    const int rt   = blockIdx.x / 12, seg = blockIdx.x % 12;
    const int rowbase = rt * 16;

    const float *W, *Bias; int ldw, cb, mode, g = 0, qi = 0;
    if (seg < 2)      { mode = 0; g = seg;     W = WKw; Bias = WKb; ldw = DIM*NG; cb = g*DIM; }
    else if (seg < 4) { mode = 1; g = seg - 2; W = WVw; Bias = WVb; ldw = DIM*NG; cb = g*DIM; }
    else              { mode = 2; qi = seg - 4; W = WQw; Bias = WQb; ldw = KD;    cb = qi*DIM; }

    f32x8 acc[4];
#pragma unroll
    for (int j = 0; j < 4; ++j)
#pragma unroll
        for (int e = 0; e < 8; ++e) acc[j][e] = 0.f;

    for (int kk = 0; kk < DIM/32; ++kk) {
        const int kb = kk * 32;
        // stage A (x tile, 16x32): one float4 + one packed b64 per thread
        {
            const int i = t*4, r = i >> 5, c = i & 31;
            const f32x4 w = *(const f32x4*)&x[(size_t)(rowbase + r)*DIM + kb + c];
            u32x2 p; p[0] = pk2(w[0], w[1]); p[1] = pk2(w[2], w[3]);
            *(u32x2*)&Alds[r*32 + c] = p;
        }
        stage_B(Blds, W, kb, ldw, cb, t);
        __syncthreads();
        Frag af;
        af.q[0] = *(const u32x4*)&Alds[ln16*32 + half*8];
        af.q[1] = *(const u32x4*)&Alds[ln16*32 + 16 + half*8];
#pragma unroll
        for (int j = 0; j < 4; ++j) {
            const int nt = wv*4 + j;
            Frag bfr;
            bfr.q[0] = *(const u32x4*)&Blds[(nt*16 + ln16)*32 + half*8];
            bfr.q[1] = *(const u32x4*)&Blds[(nt*16 + ln16)*32 + 16 + half*8];
            acc[j] = wmma_bf16(af, bfr, acc[j]);
        }
        __syncthreads();
    }

    // bias + park tile in LDS (C layout: elem r -> row r+half*8, col nt*16+ln16)
#pragma unroll
    for (int j = 0; j < 4; ++j) {
        const int col = (wv*4 + j)*16 + ln16;
        const float bv = Bias[cb + col];
#pragma unroll
        for (int r = 0; r < 8; ++r)
            Ct[(r + half*8)*256 + col] = acc[j][r] + bv;
    }
    __syncthreads();

    if (mode == 1) {
        // V: transpose-store (head_dim, seq); 8 consecutive s per (col,b)
        // pack into one b128 store each.
#pragma unroll
        for (int u2 = 0; u2 < 4; ++u2) {
            const int task = t + u2*128;          // 0..511
            const int col = task >> 1, bb = task & 1;
            const int s0 = rowbase >> 1;          // 8 consecutive s values
            u32x4 p;
#pragma unroll
            for (int j = 0; j < 4; ++j) {
                const float lo = Ct[((2*j)*2   + bb)*256 + col];
                const float hi = Ct[((2*j + 1)*2 + bb)*256 + col];
                p[j] = pk2(lo, hi);
            }
            *(u32x4*)&Vt[((size_t)((bb*NG + g)*DIM) + col)*SEQ + s0] = p;
        }
    } else {
        // K/Q: LayerNorm over 256 then bf16 cast, vectorized
        const int row = t >> 3, part = t & 7, c0 = part*32;
        const float* crow = &Ct[row*256 + c0];
        float s1 = 0.f, s2 = 0.f;
#pragma unroll
        for (int i = 0; i < 32; i += 4) {
            const f32x4 v = *(const f32x4*)&crow[i];
#pragma unroll
            for (int e = 0; e < 4; ++e) { s1 += v[e]; s2 += v[e]*v[e]; }
        }
        psum[row*8 + part] = s1; psq[row*8 + part] = s2;
        __syncthreads();
        if (part == 0) {
            float a = 0.f, b2 = 0.f;
#pragma unroll
            for (int i = 0; i < 8; ++i) { a += psum[row*8 + i]; b2 += psq[row*8 + i]; }
            const float mean = a * (1.f/256.f);
            const float var  = b2 * (1.f/256.f) - mean*mean;
            rmean[row] = mean; rrstd[row] = rsqrtf(var + 1e-5f);
        }
        __syncthreads();
        const float mean = rmean[row], rstd = rrstd[row];
        const int tok = rowbase + row;
        bf16_t* dst = (mode == 0) ? (Kn + (size_t)(tok*NG + g)*DIM)
                                  : (Qn + (size_t)(tok*NG*NSUB + qi)*DIM);
#pragma unroll
        for (int i = 0; i < 32; i += 4) {
            const f32x4 v  = *(const f32x4*)&crow[i];
            const f32x4 gg = *(const f32x4*)&lng[c0 + i];
            const f32x4 bb = *(const f32x4*)&lnb[c0 + i];
            float y[4];
#pragma unroll
            for (int e = 0; e < 4; ++e) y[e] = (v[e] - mean)*rstd*gg[e] + bb[e];
            u32x2 p; p[0] = pk2(y[0], y[1]); p[1] = pk2(y[2], y[3]);
            *(u32x2*)&dst[c0 + i] = p;
        }
    }
}

// ---------------------------------------------------------------------------
// Kernel 2: causal flash attention per head. Block = 4 waves; wave owns 16
// query rows. S = Q K^T via 16 WMMA / 32-key step; online softmax with
// 16-lane shfl_xor butterflies; P re-shaped through LDS; O += P V via 16
// WMMA / step against pre-transposed V. Epilogue bounced through LDS for
// coalesced b128 stores.
// ---------------------------------------------------------------------------
__global__ __launch_bounds__(128)
void flash_attn(const bf16_t* __restrict__ Qn, const bf16_t* __restrict__ Kn,
                const bf16_t* __restrict__ Vt, bf16_t* __restrict__ Ob)
{
    __shared__ __align__(16) bf16_t Pl[4][16*32];

    const int t = threadIdx.x, lane = t & 31, wv = t >> 5;
    const int half = lane >> 4, ln16 = lane & 15;
    const int head = blockIdx.x >> 5, qt = blockIdx.x & 31;
    const int b = head >> 3, g = (head >> 2) & 1, sub = head & 3;
    const int hq = g*NSUB + sub;
    const int q0 = qt*64 + wv*16;
    const bf16_t* Vh = Vt + (size_t)((b*NG + g)*DIM)*SEQ;
    const float scale = 0.0625f;   // 1/sqrt(256)

    f32x8 o[16];
#pragma unroll
    for (int nt = 0; nt < 16; ++nt)
#pragma unroll
        for (int e = 0; e < 8; ++e) o[nt][e] = 0.f;
    float marr[8], larr[8];
#pragma unroll
    for (int r = 0; r < 8; ++r) { marr[r] = -1e30f; larr[r] = 0.f; }

    const int nsteps = qt*2 + 2;   // uniform across waves -> barriers legal
    for (int step = 0; step < nsteps; ++step) {
        const int kb = step*32;
        if (step + 1 < nsteps) {   // hide L2 latency: global_prefetch_b8
            __builtin_prefetch(Kn + ((size_t)((kb + 32 + ln16)*BSZ + b)*NG + g)*DIM, 0, 1);
            __builtin_prefetch(Vh + (size_t)(lane*8)*SEQ + kb + 32, 0, 1);
        }
        f32x8 s0, s1;
#pragma unroll
        for (int e = 0; e < 8; ++e) { s0[e] = 0.f; s1[e] = 0.f; }

#pragma unroll
        for (int dk = 0; dk < DIM/32; ++dk) {
            const int db = dk*32;
            Frag qa, k0, k1;
            const bf16_t* qp  = Qn + ((size_t)((q0 + ln16)*BSZ + b)*NG*NSUB + hq)*DIM + db + half*8;
            qa.q[0] = *(const u32x4*)qp;        qa.q[1] = *(const u32x4*)(qp + 16);
            const bf16_t* kp0 = Kn + ((size_t)((kb + ln16)*BSZ + b)*NG + g)*DIM + db + half*8;
            k0.q[0] = *(const u32x4*)kp0;       k0.q[1] = *(const u32x4*)(kp0 + 16);
            const bf16_t* kp1 = Kn + ((size_t)((kb + 16 + ln16)*BSZ + b)*NG + g)*DIM + db + half*8;
            k1.q[0] = *(const u32x4*)kp1;       k1.q[1] = *(const u32x4*)(kp1 + 16);
            s0 = wmma_bf16(qa, k0, s0);
            s1 = wmma_bf16(qa, k1, s1);
        }

        // scale + causal mask (only boundary steps actually branch)
        const bool need_mask = (kb + 31) > q0;
#pragma unroll
        for (int r = 0; r < 8; ++r) {
            const int row = q0 + r + half*8;
            float v0 = s0[r]*scale, v1 = s1[r]*scale;
            if (need_mask) {
                if (kb + ln16 > row)       v0 = -1e30f;
                if (kb + 16 + ln16 > row)  v1 = -1e30f;
            }
            s0[r] = v0; s1[r] = v1;
        }

        // online softmax: row stats replicated across each 16-lane half
        float alpha[8];
#pragma unroll
        for (int r = 0; r < 8; ++r) {
            float mx = fmaxf(s0[r], s1[r]);
#pragma unroll
            for (int off = 1; off < 16; off <<= 1)
                mx = fmaxf(mx, __shfl_xor(mx, off, 32));
            const float mn = fmaxf(marr[r], mx);
            alpha[r] = __expf(marr[r] - mn);
            marr[r] = mn;
            const float p0 = __expf(s0[r] - mn), p1 = __expf(s1[r] - mn);
            s0[r] = p0; s1[r] = p1;
            float rs = p0 + p1;
#pragma unroll
            for (int off = 1; off < 16; off <<= 1)
                rs += __shfl_xor(rs, off, 32);
            larr[r] = larr[r]*alpha[r] + rs;
        }
#pragma unroll
        for (int nt = 0; nt < 16; ++nt)
#pragma unroll
            for (int r = 0; r < 8; ++r) o[nt][r] *= alpha[r];

        // C-layout -> A-layout reshape of P through LDS
#pragma unroll
        for (int r = 0; r < 8; ++r) {
            const int row = r + half*8;
            Pl[wv][row*32 + ln16]      = f2bf(s0[r]);
            Pl[wv][row*32 + 16 + ln16] = f2bf(s1[r]);
        }
        __syncthreads();
        Frag pf;
        pf.q[0] = *(const u32x4*)&Pl[wv][ln16*32 + half*8];
        pf.q[1] = *(const u32x4*)&Pl[wv][ln16*32 + 16 + half*8];
#pragma unroll
        for (int nt = 0; nt < 16; ++nt) {
            Frag vf;
            const bf16_t* vp = Vh + (size_t)(nt*16 + ln16)*SEQ + kb + half*8;
            vf.q[0] = *(const u32x4*)vp;  vf.q[1] = *(const u32x4*)(vp + 16);
            o[nt] = wmma_bf16(pf, vf, o[nt]);
        }
        __syncthreads();
    }

    // Epilogue: bounce O through Pl (2 rows per pass) for coalesced stores.
    bf16_t* plw = &Pl[wv][0];           // 512 halves = 2 full 256-wide rows
    for (int r = 0; r < 8; ++r) {       // uniform across waves
        const float inv = 1.f / larr[r];
#pragma unroll
        for (int nt = 0; nt < 16; ++nt)
            plw[half*256 + nt*16 + ln16] = f2bf(o[nt][r] * inv);
        __syncthreads();
        const int rowslot = lane >> 4;                 // 0 or 1
        const int q = q0 + r + rowslot*8;
        const int d0 = (lane & 15)*16;
        const u32x4 p0 = *(const u32x4*)&plw[rowslot*256 + d0];
        const u32x4 p1 = *(const u32x4*)&plw[rowslot*256 + d0 + 8];
        bf16_t* op = Ob + ((size_t)(q*BSZ + b)*NG*NSUB + hq)*DIM + d0;
        *(u32x4*)op       = p0;
        *(u32x4*)(op + 8) = p1;
        __syncthreads();
    }
}

// ---------------------------------------------------------------------------
// Kernel 3: output projection (4096x2048) x (2048x256) + bias, fp32 out.
// ---------------------------------------------------------------------------
__global__ __launch_bounds__(128)
void out_proj(const bf16_t* __restrict__ Ob, const float* __restrict__ WOw,
              const float* __restrict__ WOb, float* __restrict__ out)
{
    __shared__ __align__(16) bf16_t Blds[256*32];
    __shared__ __align__(16) float  Cout[16*256];
    const int t = threadIdx.x, lane = t & 31, wv = t >> 5;
    const int half = lane >> 4, ln16 = lane & 15;
    const int rowbase = blockIdx.x * 16;

    f32x8 acc[4];
#pragma unroll
    for (int j = 0; j < 4; ++j)
#pragma unroll
        for (int e = 0; e < 8; ++e) acc[j][e] = 0.f;

    for (int kk = 0; kk < KD/32; ++kk) {
        const int kb = kk*32;
        stage_B(Blds, WOw, kb, DIM, 0, t);
        __syncthreads();
        Frag af;
        const bf16_t* ap = Ob + (size_t)(rowbase + ln16)*KD + kb + half*8;
        af.q[0] = *(const u32x4*)ap;  af.q[1] = *(const u32x4*)(ap + 16);
#pragma unroll
        for (int j = 0; j < 4; ++j) {
            const int nt = wv*4 + j;
            Frag bfr;
            bfr.q[0] = *(const u32x4*)&Blds[(nt*16 + ln16)*32 + half*8];
            bfr.q[1] = *(const u32x4*)&Blds[(nt*16 + ln16)*32 + 16 + half*8];
            acc[j] = wmma_bf16(af, bfr, acc[j]);
        }
        __syncthreads();
    }
    // park in LDS, then coalesced float4 stores
#pragma unroll
    for (int j = 0; j < 4; ++j) {
        const int col = (wv*4 + j)*16 + ln16;
        const float bv = WOb[col];
#pragma unroll
        for (int r = 0; r < 8; ++r)
            Cout[(r + half*8)*256 + col] = acc[j][r] + bv;
    }
    __syncthreads();
    {
        const int row = t >> 3, c0 = (t & 7)*32;
#pragma unroll
        for (int i = 0; i < 32; i += 4) {
            const f32x4 v = *(const f32x4*)&Cout[row*256 + c0 + i];
            *(f32x4*)&out[(size_t)(rowbase + row)*DIM + c0 + i] = v;
        }
    }
}

// ---------------------------------------------------------------------------
extern "C" void kernel_launch(void* const* d_in, const int* in_sizes, int n_in,
                              void* d_out, int out_size, void* d_ws, size_t ws_size,
                              hipStream_t stream)
{
    (void)in_sizes; (void)n_in; (void)out_size; (void)ws_size;
    const float* x   = (const float*)d_in[0];
    const float* WKw = (const float*)d_in[1];
    const float* WKb = (const float*)d_in[2];
    const float* WVw = (const float*)d_in[3];
    const float* WVb = (const float*)d_in[4];
    const float* WQw = (const float*)d_in[5];
    const float* WQb = (const float*)d_in[6];
    const float* WOw = (const float*)d_in[7];
    const float* WOb = (const float*)d_in[8];
    const float* lng = (const float*)d_in[9];
    const float* lnb = (const float*)d_in[10];

    char* ws = (char*)d_ws;
    bf16_t* Kn = (bf16_t*)(ws);                              //  4 MB
    bf16_t* Qn = (bf16_t*)(ws + (size_t) 4*1024*1024);       // 16 MB
    bf16_t* Vt = (bf16_t*)(ws + (size_t)20*1024*1024);       //  4 MB
    bf16_t* Ob = (bf16_t*)(ws + (size_t)24*1024*1024);       // 16 MB

    proj_qkv_ln<<<dim3((NTOK/16)*12), 128, 0, stream>>>(
        x, WKw, WKb, WVw, WVb, WQw, WQb, lng, lnb, Kn, Qn, Vt);
    flash_attn<<<dim3(BSZ*NG*NSUB*(SEQ/64)), 128, 0, stream>>>(Qn, Kn, Vt, Ob);
    out_proj<<<dim3(NTOK/16), 128, 0, stream>>>(Ob, WOw, WOb, (float*)d_out);
}
